// DotProductAttention_27023934227141
// MI455X (gfx1250) — matrix operand order repaired and checked by reading, not compiled, not run on hardware
//
#include <hip/hip_runtime.h>

typedef __attribute__((ext_vector_type(16))) _Float16 v16h;
typedef __attribute__((ext_vector_type(8)))  _Float16 v8h;
typedef __attribute__((ext_vector_type(4)))  _Float16 v4h;
typedef __attribute__((ext_vector_type(8)))  float    v8f;
typedef __attribute__((ext_vector_type(4)))  float    v4f;

#define LDP 72   // padded LDS row stride (f16): 144B = 36-bank step -> conflict-free b128 frag loads

#define WMMA_F16(A, B, C) \
  __builtin_amdgcn_wmma_f32_16x16x32_f16(false, (A), false, (B), (short)0, (C), false, false)

__device__ __forceinline__ v16h ld_frag(const _Float16* p0, const _Float16* p1) {
  v8h lo = *(const v8h*)p0;   // ds_load_b128
  v8h hi = *(const v8h*)p1;   // ds_load_b128
  return __builtin_shufflevector(lo, hi, 0,1,2,3,4,5,6,7,8,9,10,11,12,13,14,15);
}

__global__ __launch_bounds__(128)
void attn_fwd_f16wmma(const float* __restrict__ Qg_, const float* __restrict__ Kg_,
                      const float* __restrict__ Vg_, float* __restrict__ Og_)
{
  constexpr int S = 1024, D = 64;
  constexpr int NKB = S / 64;
  const int qt   = blockIdx.x;
  const int bh   = blockIdx.y;
  const int tid  = threadIdx.x;
  const int wave = tid >> 5;
  const int lane = tid & 31;
  const int half = lane >> 4;
  const int l16  = lane & 15;

  __shared__ __align__(16) _Float16 Qs[64 * LDP];
  __shared__ __align__(16) _Float16 Ks[2][64 * LDP];   // double-buffered K [key][d]
  __shared__ __align__(16) _Float16 Vt[2][64 * LDP];   // double-buffered V^T [d][key]
  __shared__ __align__(16) _Float16 Ps[4 * 16 * LDP];  // per-wave P strip

  const float qscale = 0.125f * 1.44269504088896341f;  // 1/sqrt(64) * log2(e)

  const float* Qg = Qg_ + (size_t)(bh * S + qt * 64) * D;
  const float* Kg = Kg_ + (size_t)bh * S * D;
  const float* Vg = Vg_ + (size_t)bh * S * D;

  // ---- stage Q tile (64x64), float4 loads -> v4h LDS stores ----
  #pragma unroll
  for (int it = 0; it < 8; ++it) {
    int f = tid + it * 128;            // float4 id (1024 per tile)
    int r = f >> 4, c = (f & 15) * 4;
    v4f q = *(const v4f*)(Qg + r * D + c);
    v4h h;
    #pragma unroll
    for (int u = 0; u < 4; ++u) h[u] = (_Float16)(q[u] * qscale);
    *(v4h*)&Qs[r * LDP + c] = h;
  }
  __syncthreads();

  const int qrow = wave * 16 + l16;
  v16h qa0 = ld_frag(&Qs[qrow * LDP +  0 + 8 * half], &Qs[qrow * LDP + 16 + 8 * half]);
  v16h qa1 = ld_frag(&Qs[qrow * LDP + 32 + 8 * half], &Qs[qrow * LDP + 48 + 8 * half]);

  // V staging mapping: thread owns one d-column (kd) and key-quads; write is a b64 row in Vt
  const int kd  = tid & 63;
  const int kq0 = tid >> 6;

  // ---- preload block 0 into buffer 0 ----
  {
    #pragma unroll
    for (int it = 0; it < 8; ++it) {
      int f = tid + it * 128; int r = f >> 4, c = (f & 15) * 4;
      v4f k = *(const v4f*)(Kg + r * D + c);
      v4h h;
      #pragma unroll
      for (int u = 0; u < 4; ++u) h[u] = (_Float16)k[u];
      *(v4h*)&Ks[0][r * LDP + c] = h;
    }
    #pragma unroll
    for (int it = 0; it < 8; ++it) {
      int q = it * 2 + kq0;            // key quad 0..15
      v4h h;
      #pragma unroll
      for (int u = 0; u < 4; ++u) h[u] = (_Float16)Vg[(q * 4 + u) * D + kd];
      *(v4h*)&Vt[0][kd * LDP + q * 4] = h;
    }
  }
  __syncthreads();

  v8f   o[4];
  float m[8], l[8];
  #pragma unroll
  for (int j = 0; j < 4; ++j)
    #pragma unroll
    for (int i = 0; i < 8; ++i) o[j][i] = 0.f;
  #pragma unroll
  for (int i = 0; i < 8; ++i) { m[i] = -__builtin_inff(); l[i] = 0.f; }

  _Float16* Pw = Ps + wave * 16 * LDP;

  for (int kb = 0; kb < NKB; ++kb) {
    const int  cur  = kb & 1, nxt = cur ^ 1;
    const bool more = (kb + 1 < NKB);

    // ---- issue next block's global loads NOW (latency hidden behind WMMAs) ----
    v4f   kreg[8];
    float vreg[8][4];
    if (more) {
      const float* Kblk = Kg + (size_t)(kb + 1) * 64 * D;
      const float* Vblk = Vg + (size_t)(kb + 1) * 64 * D;
      #pragma unroll
      for (int it = 0; it < 8; ++it) {
        int f = tid + it * 128; int r = f >> 4, c = (f & 15) * 4;
        kreg[it] = *(const v4f*)(Kblk + r * D + c);
      }
      #pragma unroll
      for (int it = 0; it < 8; ++it) {
        int q = it * 2 + kq0;
        #pragma unroll
        for (int u = 0; u < 4; ++u) vreg[it][u] = Vblk[(q * 4 + u) * D + kd];
      }
    }

    // ---- S = Q K^T : 8 WMMAs ----
    v8f s[4];
    #pragma unroll
    for (int j = 0; j < 4; ++j) {
      const int kr = j * 16 + l16;
      v16h b0 = ld_frag(&Ks[cur][kr * LDP +  0 + 8 * half], &Ks[cur][kr * LDP + 16 + 8 * half]);
      v16h b1 = ld_frag(&Ks[cur][kr * LDP + 32 + 8 * half], &Ks[cur][kr * LDP + 48 + 8 * half]);
      v8f acc;
      #pragma unroll
      for (int i = 0; i < 8; ++i) acc[i] = 0.f;
      acc = WMMA_F16(qa0, b0, acc);
      acc = WMMA_F16(qa1, b1, acc);
      s[j] = acc;
    }

    // ---- online softmax: batched cross-lane reductions (8 rows per xor step) ----
    float mx[8];
    #pragma unroll
    for (int i = 0; i < 8; ++i)
      mx[i] = fmaxf(fmaxf(s[0][i], s[1][i]), fmaxf(s[2][i], s[3][i]));
    #pragma unroll
    for (int st = 1; st < 16; st <<= 1) {
      #pragma unroll
      for (int i = 0; i < 8; ++i) mx[i] = fmaxf(mx[i], __shfl_xor(mx[i], st));
    }
    float alpha[8], rs[8];
    #pragma unroll
    for (int i = 0; i < 8; ++i) {
      float mn = fmaxf(m[i], mx[i]);
      alpha[i] = exp2f(m[i] - mn);
      m[i] = mn;
      float r0 = 0.f;
      #pragma unroll
      for (int j = 0; j < 4; ++j) { float p = exp2f(s[j][i] - mn); s[j][i] = p; r0 += p; }
      rs[i] = r0;
    }
    #pragma unroll
    for (int st = 1; st < 16; st <<= 1) {
      #pragma unroll
      for (int i = 0; i < 8; ++i) rs[i] += __shfl_xor(rs[i], st);
    }
    #pragma unroll
    for (int i = 0; i < 8; ++i) {
      l[i] = l[i] * alpha[i] + rs[i];
      #pragma unroll
      for (int j = 0; j < 4; ++j) o[j][i] *= alpha[i];
    }

    // ---- spill P as f16 (C-layout -> A-layout via LDS) ----
    #pragma unroll
    for (int i = 0; i < 8; ++i)
      #pragma unroll
      for (int j = 0; j < 4; ++j)
        Pw[(8 * half + i) * LDP + j * 16 + l16] = (_Float16)s[j][i];
    __builtin_amdgcn_wave_barrier();

    // ---- stage next block into the alternate buffers (waits on global loads here) ----
    if (more) {
      #pragma unroll
      for (int it = 0; it < 8; ++it) {
        int f = tid + it * 128; int r = f >> 4, c = (f & 15) * 4;
        v4h h;
        #pragma unroll
        for (int u = 0; u < 4; ++u) h[u] = (_Float16)kreg[it][u];
        *(v4h*)&Ks[nxt][r * LDP + c] = h;
      }
      #pragma unroll
      for (int it = 0; it < 8; ++it) {
        int q = it * 2 + kq0;
        v4h h;
        #pragma unroll
        for (int u = 0; u < 4; ++u) h[u] = (_Float16)vreg[it][u];
        *(v4h*)&Vt[nxt][kd * LDP + q * 4] = h;
      }
    }

    // ---- O += P V : 8 WMMAs ----
    v16h pa0 = ld_frag(&Pw[l16 * LDP +  0 + 8 * half], &Pw[l16 * LDP + 16 + 8 * half]);
    v16h pa1 = ld_frag(&Pw[l16 * LDP + 32 + 8 * half], &Pw[l16 * LDP + 48 + 8 * half]);
    #pragma unroll
    for (int j = 0; j < 4; ++j) {
      const int vr = j * 16 + l16;
      v16h b0 = ld_frag(&Vt[cur][vr * LDP +  0 + 8 * half], &Vt[cur][vr * LDP + 16 + 8 * half]);
      v16h b1 = ld_frag(&Vt[cur][vr * LDP + 32 + 8 * half], &Vt[cur][vr * LDP + 48 + 8 * half]);
      o[j] = WMMA_F16(pa0, b0, o[j]);
      o[j] = WMMA_F16(pa1, b1, o[j]);
    }
    __syncthreads();   // next-buffer writes visible; cur buffer safe to overwrite next iter
  }

  // ---- normalize and store O ----
  float* Og = Og_ + (size_t)(bh * S + qt * 64 + wave * 16) * D;
  #pragma unroll
  for (int i = 0; i < 8; ++i) {
    float inv = 1.f / l[i];
    int r = 8 * half + i;
    #pragma unroll
    for (int j = 0; j < 4; ++j)
      Og[r * D + j * 16 + l16] = o[j][i] * inv;
  }
}

extern "C" void kernel_launch(void* const* d_in, const int* in_sizes, int n_in,
                              void* d_out, int out_size, void* d_ws, size_t ws_size,
                              hipStream_t stream) {
  const float* Q = (const float*)d_in[0];
  const float* K = (const float*)d_in[1];
  const float* V = (const float*)d_in[2];
  float* Out = (float*)d_out;
  dim3 grid(16, 64);   // x: S/64 query tiles, y: B*H heads
  attn_fwd_f16wmma<<<grid, dim3(128), 0, stream>>>(Q, K, V, Out);
}
